// MambaAttn_14190571946258
// MI455X (gfx1250) — matrix operand order repaired and checked by reading, not compiled, not run on hardware
//
#include <hip/hip_runtime.h>
#include <cmath>

typedef _Float16 v16h __attribute__((ext_vector_type(16)));
typedef _Float16 v8h  __attribute__((ext_vector_type(8)));
typedef float    v8f  __attribute__((ext_vector_type(8)));

#define LDA 72            // padded row stride (halves) for LDS tiles
#define CHUNK 64          // scan chunk length
#define NCHUNK 256        // 16384 / CHUNK

static __device__ __forceinline__ float rcp_fast(float v) {
    return __builtin_amdgcn_rcpf(v);           // single v_rcp_f32
}
static __device__ __forceinline__ float siluf(float v) {
    return v * rcp_fast(1.f + __expf(-v));
}
static __device__ __forceinline__ float sigmf(float v) {
    return rcp_fast(1.f + __expf(-v));
}
static __device__ __forceinline__ float softplusf(float v) {
    return (v > 20.f) ? v : log1pf(__expf(v));
}

// ---- WMMA fragment helpers (CDNA5 16x16x32 f16, wave32) -------------------
// A tile in LDS: row-major M x K, rows padded to LDA halves.
// Lane (m = lane&15, kh = lane>>4) holds K = kh*8 + 0..7 and 16 + kh*8 + 0..7.
static __device__ __forceinline__ v16h frag_a(const _Float16* As, int row0, int kk, int lane) {
    const int m = lane & 15, kh = lane >> 4;
    const _Float16* p = As + (row0 + m) * LDA + kk + kh * 8;
    v8h lo = *(const v8h*)p;
    v8h hi = *(const v8h*)(p + 16);
    return __builtin_shufflevector(lo, hi, 0,1,2,3,4,5,6,7,8,9,10,11,12,13,14,15);
}
// B tile staged as weight rows [N][K] row-major (i.e. B^T): lane n = lane&15,
// holds K = kh*8 + 0..7 and 16 + kh*8 + 0..7.
static __device__ __forceinline__ v16h frag_b(const _Float16* Ws, int n0, int kk, int lane) {
    const int n = lane & 15, kh = lane >> 4;
    const _Float16* p = Ws + (n0 + n) * LDA + kk + kh * 8;
    v8h lo = *(const v8h*)p;
    v8h hi = *(const v8h*)(p + 16);
    return __builtin_shufflevector(lo, hi, 0,1,2,3,4,5,6,7,8,9,10,11,12,13,14,15);
}
static __device__ __forceinline__ v8f wmma32(v16h a, v16h b, v8f c) {
    return __builtin_amdgcn_wmma_f32_16x16x32_f16(false, a, false, b, (short)0, c, false, false);
}

// ===========================================================================
// K1: xz = x @ m_in_w^T (N=128) fused with x_mask = silu(x @ W_toMask^T) (N=64)
// M = 32768, K = 64, combined N = 192.
// Waves 0..3 own the xz tiles (ct 0..7); waves 4..7 own the x_mask tiles
// (ct 8..11) -> each arm of the wave-id branch is itself branch-free.
__global__ void k1_inproj(const float* __restrict__ x, const float* __restrict__ w_in,
                          const float* __restrict__ w_mask,
                          float* __restrict__ xz, float* __restrict__ xmask) {
    __shared__ __align__(16) _Float16 Ash[64 * LDA];
    __shared__ __align__(16) _Float16 Wsh[192 * LDA];
    const int tid = threadIdx.x;
    const int row0 = blockIdx.x * 64;
    for (int i = tid; i < 192 * 64; i += 256) {
        int n = i >> 6, k = i & 63;
        float w = (n < 128) ? w_in[n * 64 + k] : w_mask[(n - 128) * 64 + k];
        Wsh[n * LDA + k] = (_Float16)w;
    }
    for (int i = tid; i < 64 * 64; i += 256) {
        int r = i >> 6, k = i & 63;
        Ash[r * LDA + k] = (_Float16)x[(long)(row0 + r) * 64 + k];
    }
    __syncthreads();
    const int wave = tid >> 5, lane = tid & 31;
    const int rt = wave & 3;
    const v16h a0 = frag_a(Ash, rt * 16, 0, lane);
    const v16h a1 = frag_a(Ash, rt * 16, 32, lane);
    const int  nl = lane & 15;
    const long mbase = (long)row0 + rt * 16 + ((lane >> 4) << 3);
    if (wave < 4) {                    // one branch per kernel, then uniform work
#pragma unroll
        for (int ct = 0; ct < 8; ++ct) {
            v8f c = {};
            c = wmma32(a0, frag_b(Wsh, ct * 16, 0, lane),  c);
            c = wmma32(a1, frag_b(Wsh, ct * 16, 32, lane), c);
            int n = ct * 16 + nl;
#pragma unroll
            for (int v = 0; v < 8; ++v) xz[(mbase + v) * 128 + n] = c[v];
        }
    } else {
#pragma unroll
        for (int ct = 8; ct < 12; ++ct) {
            v8f c = {};
            c = wmma32(a0, frag_b(Wsh, ct * 16, 0, lane),  c);
            c = wmma32(a1, frag_b(Wsh, ct * 16, 32, lane), c);
            int n = ct * 16 - 128 + nl;
#pragma unroll
            for (int v = 0; v < 8; ++v) xmask[(mbase + v) * 64 + n] = siluf(c[v]);
        }
    }
}

// K0: wdt_eff[dir][d][k] = sum_r dt_w[d][r] * xproj[r][k]   (fold dt proj)
__global__ void k0_wdt(const float* __restrict__ dtw_f, const float* __restrict__ xp_f,
                       const float* __restrict__ dtw_b, const float* __restrict__ xp_b,
                       float* __restrict__ wdt) {
    int idx = blockIdx.x * blockDim.x + threadIdx.x;   // 8192
    int k = idx & 63, d = (idx >> 6) & 63, dir = idx >> 12;
    const float* dtw = dir ? dtw_b : dtw_f;
    const float* xp  = dir ? xp_b  : xp_f;
    float s = 0.f;
#pragma unroll
    for (int r = 0; r < 4; ++r) s += dtw[d * 4 + r] * xp[r * 64 + k];
    wdt[idx] = s;
}

// K2a: depthwise causal conv (DCONV=4) + silu, both directions (bw reads flipped)
__global__ void k2a_conv(const float* __restrict__ xz,
                         const float* __restrict__ cw_f, const float* __restrict__ cb_f,
                         const float* __restrict__ cw_b, const float* __restrict__ cb_b,
                         float* __restrict__ xc) {
    int idx = blockIdx.x * blockDim.x + threadIdx.x;   // 4194304
    int d = idx & 63;
    int l = (idx >> 6) & 16383;
    int b = (idx >> 20) & 1;
    int dir = idx >> 21;
    const float* cw = dir ? cw_b : cw_f;
    const float* cb = dir ? cb_b : cb_f;
    float acc = cb[d];
#pragma unroll
    for (int k = 0; k < 4; ++k) {
        int ls = l - 3 + k;
        if (ls >= 0) {
            int src = dir ? 16383 - ls : ls;
            acc += cw[d * 4 + k] * xz[((long)b * 16384 + src) * 128 + d];
        }
    }
    xc[idx] = siluf(acc);
}

// K2b: per-direction GEMM [32768 x 64] @ [64 x 80]^T.
// N layout: 0..63 -> dt (softplus), 64..79 -> BC (B states 0..7, C states 8..15).
// Waves 0..3 own the dt tiles, waves 4..7 own the BC tile.
__global__ void k2b_proj(const float* __restrict__ xc, const float* __restrict__ wdt,
                         const float* __restrict__ xp_f, const float* __restrict__ xp_b,
                         const float* __restrict__ dtb_f, const float* __restrict__ dtb_b,
                         float* __restrict__ dt, float* __restrict__ BC) {
    __shared__ __align__(16) _Float16 Ash[64 * LDA];
    __shared__ __align__(16) _Float16 Wsh[80 * LDA];
    const int tid = threadIdx.x;
    const int dir = blockIdx.y;
    const int row0 = blockIdx.x * 64;                   // within this direction
    const float* xcd = xc + (long)dir * 2097152;
    const float* xp  = dir ? xp_b  : xp_f;
    const float* dtb = dir ? dtb_b : dtb_f;
    for (int i = tid; i < 80 * 64; i += 256) {
        int n = i >> 6, k = i & 63;
        float w = (n < 64) ? wdt[dir * 4096 + n * 64 + k] : xp[(4 + n - 64) * 64 + k];
        Wsh[n * LDA + k] = (_Float16)w;
    }
    for (int i = tid; i < 4096; i += 256) {
        int r = i >> 6, k = i & 63;
        Ash[r * LDA + k] = (_Float16)xcd[(long)(row0 + r) * 64 + k];
    }
    __syncthreads();
    const int wave = tid >> 5, lane = tid & 31;
    const int rt = wave & 3;
    const v16h a0 = frag_a(Ash, rt * 16, 0, lane);
    const v16h a1 = frag_a(Ash, rt * 16, 32, lane);
    const int  nl = lane & 15;
    const long mbase = (long)dir * 32768 + row0 + rt * 16 + ((lane >> 4) << 3);
    if (wave < 4) {                         // dt tiles: ct 0..3
#pragma unroll
        for (int ct = 0; ct < 4; ++ct) {
            v8f c = {};
            c = wmma32(a0, frag_b(Wsh, ct * 16, 0, lane),  c);
            c = wmma32(a1, frag_b(Wsh, ct * 16, 32, lane), c);
            int n = ct * 16 + nl;
            float bn = dtb[n];
#pragma unroll
            for (int v = 0; v < 8; ++v) dt[(mbase + v) * 64 + n] = softplusf(c[v] + bn);
        }
    } else {                                // BC tile: ct == 4
        v8f c = {};
        c = wmma32(a0, frag_b(Wsh, 64, 0, lane),  c);
        c = wmma32(a1, frag_b(Wsh, 64, 32, lane), c);
#pragma unroll
        for (int v = 0; v < 8; ++v) BC[(mbase + v) * 16 + nl] = c[v];
    }
}

// K3: chunk-local scan aggregates (A,B) per (dir,b,chunk,d,n)
__global__ void k3_scan1(const float* __restrict__ dt, const float* __restrict__ xc,
                         const float* __restrict__ BC,
                         const float* __restrict__ Alog_f, const float* __restrict__ Alog_b,
                         float* __restrict__ aggA, float* __restrict__ aggB) {
    __shared__ float dts[CHUNK * 64];
    __shared__ float xcs[CHUNK * 64];
    __shared__ float bcs[CHUNK * 16];
    const int chunk = blockIdx.x, b = blockIdx.y, dir = blockIdx.z;
    const int tid = threadIdx.x;                        // 64 threads == channel d
    long base = ((long)(dir * 2 + b) * 16384 + (long)chunk * CHUNK);
    const float* dtp = dt + base * 64;
    const float* xcp = xc + base * 64;
    const float* bcp = BC + base * 16;
    for (int i = tid; i < CHUNK * 64; i += 64) { dts[i] = dtp[i]; xcs[i] = xcp[i]; }
    for (int i = tid; i < CHUNK * 16; i += 64) bcs[i] = bcp[i];
    __syncthreads();
    const float* Alog = dir ? Alog_b : Alog_f;
    float A[8], aA[8], aB[8];
#pragma unroll
    for (int n = 0; n < 8; ++n) { A[n] = -__expf(Alog[tid * 8 + n]); aA[n] = 1.f; aB[n] = 0.f; }
    for (int l = 0; l < CHUNK; ++l) {
        float dtv = dts[l * 64 + tid];
        float dx  = dtv * xcs[l * 64 + tid];
#pragma unroll
        for (int n = 0; n < 8; ++n) {
            float da = __expf(dtv * A[n]);
            aB[n] = aB[n] * da + dx * bcs[l * 16 + n];
            aA[n] *= da;
        }
    }
    long o = (((long)(dir * 2 + b) * NCHUNK + chunk) * 64 + tid) * 8;
#pragma unroll
    for (int n = 0; n < 8; ++n) { aggA[o + n] = aA[n]; aggB[o + n] = aB[n]; }
}

// K4: sequential scan of chunk aggregates -> exclusive carry per chunk
__global__ void k4_carry(const float* __restrict__ aggA, const float* __restrict__ aggB,
                         float* __restrict__ carry) {
    int idx = blockIdx.x * blockDim.x + threadIdx.x;   // 2048 states
    int n = idx & 7, d = (idx >> 3) & 63, db = idx >> 9;
    float h = 0.f;
    for (int c = 0; c < NCHUNK; ++c) {
        long o = (((long)db * NCHUNK + c) * 64 + d) * 8 + n;
        carry[o] = h;
        h = aggA[o] * h + aggB[o];
    }
}

// K5: replay with carry-in, fuse y = sum_n h*C + xc*D, gate with silu(z), un-flip bw
__global__ void k5_scan2(const float* __restrict__ dt, const float* __restrict__ xc,
                         const float* __restrict__ BC,
                         const float* __restrict__ xz, const float* __restrict__ carry,
                         const float* __restrict__ Alog_f, const float* __restrict__ Alog_b,
                         const float* __restrict__ D_f, const float* __restrict__ D_b,
                         float* __restrict__ y) {
    __shared__ float dts[CHUNK * 64];
    __shared__ float xcs[CHUNK * 64];
    __shared__ float bcs[CHUNK * 16];
    const int chunk = blockIdx.x, b = blockIdx.y, dir = blockIdx.z;
    const int tid = threadIdx.x;
    long base = ((long)(dir * 2 + b) * 16384 + (long)chunk * CHUNK);
    const float* dtp = dt + base * 64;
    const float* xcp = xc + base * 64;
    const float* bcp = BC + base * 16;
    for (int i = tid; i < CHUNK * 64; i += 64) { dts[i] = dtp[i]; xcs[i] = xcp[i]; }
    for (int i = tid; i < CHUNK * 16; i += 64) bcs[i] = bcp[i];
    __syncthreads();
    const float* Alog = dir ? Alog_b : Alog_f;
    const float  Dd   = (dir ? D_b : D_f)[tid];
    long co = (((long)(dir * 2 + b) * NCHUNK + chunk) * 64 + tid) * 8;
    float A[8], h[8];
#pragma unroll
    for (int n = 0; n < 8; ++n) { A[n] = -__expf(Alog[tid * 8 + n]); h[n] = carry[co + n]; }
    for (int l = 0; l < CHUNK; ++l) {
        float dtv = dts[l * 64 + tid];
        float xv  = xcs[l * 64 + tid];
        float dx  = dtv * xv;
        float yv  = 0.f;
#pragma unroll
        for (int n = 0; n < 8; ++n) {
            float da = __expf(dtv * A[n]);
            h[n] = h[n] * da + dx * bcs[l * 16 + n];
            yv += h[n] * bcs[l * 16 + 8 + n];
        }
        yv += xv * Dd;
        int lpos = chunk * CHUNK + l;
        int pos  = dir ? 16383 - lpos : lpos;
        float z = xz[((long)b * 16384 + pos) * 128 + 64 + tid];
        y[(((long)(dir * 2 + b)) * 16384 + pos) * 64 + tid] = yv * siluf(z);
    }
}

// K6: y = 0.5*(y_f + y_b), RMSNorm*gamma, x_mamba = y @ m_out_w^T  (WMMA)
__global__ void k6_outproj(const float* __restrict__ y, const float* __restrict__ norm_w,
                           const float* __restrict__ w_out, float* __restrict__ xmamba) {
    __shared__ float Yf[64 * 68];
    __shared__ float rsms[64];
    __shared__ __align__(16) _Float16 Ash[64 * LDA];
    __shared__ __align__(16) _Float16 Wsh[64 * LDA];
    const int tid = threadIdx.x;
    const int row0 = blockIdx.x * 64;                   // rows over (b,l)
    for (int i = tid; i < 4096; i += 256) {
        int r = i >> 6, k = i & 63;
        long g = (long)(row0 + r) * 64 + k;
        Yf[r * 68 + k] = 0.5f * (y[g] + y[g + 2097152]);
    }
    for (int i = tid; i < 4096; i += 256)
        Wsh[(i >> 6) * LDA + (i & 63)] = (_Float16)w_out[i];
    __syncthreads();
    if (tid < 64) {
        float ss = 0.f;
        for (int k = 0; k < 64; ++k) { float v = Yf[tid * 68 + k]; ss += v * v; }
        rsms[tid] = rsqrtf(ss * (1.f / 64.f) + 1e-5f);
    }
    __syncthreads();
    for (int i = tid; i < 4096; i += 256) {
        int r = i >> 6, k = i & 63;
        Ash[r * LDA + k] = (_Float16)(Yf[r * 68 + k] * rsms[r] * norm_w[k]);
    }
    __syncthreads();
    const int wave = tid >> 5, lane = tid & 31;
    const int rt = wave & 3, cg = wave >> 2;
    const v16h a0 = frag_a(Ash, rt * 16, 0, lane);
    const v16h a1 = frag_a(Ash, rt * 16, 32, lane);
    const int  nl = lane & 15;
    const long mbase = (long)row0 + rt * 16 + ((lane >> 4) << 3);
#pragma unroll
    for (int q = 0; q < 2; ++q) {
        int ct = cg * 2 + q;
        v8f c = {};
        c = wmma32(a0, frag_b(Wsh, ct * 16, 0, lane),  c);
        c = wmma32(a1, frag_b(Wsh, ct * 16, 32, lane), c);
        int n = ct * 16 + nl;
#pragma unroll
        for (int v = 0; v < 8; ++v) xmamba[(mbase + v) * 64 + n] = c[v];
    }
}

// K7: 1x1 conv over NCHW pixels (used twice): out[b,o,p] = sum_c w[o,c]*in[b,c,p] + bias[o]
__global__ void k7_conv1x1(const float* __restrict__ in, const float* __restrict__ w,
                           const float* __restrict__ bias, float* __restrict__ out) {
    __shared__ __align__(16) _Float16 Ash[64 * LDA];
    __shared__ __align__(16) _Float16 Wsh[64 * LDA];
    const int tid = threadIdx.x;
    const int b = blockIdx.y;
    const int p0 = blockIdx.x * 64;
    const float* inb = in + (long)b * 64 * 20480;
    for (int i = tid; i < 4096; i += 256) {
        int c = i >> 6, p = i & 63;
        Ash[p * LDA + c] = (_Float16)inb[(long)c * 20480 + p0 + p];   // transpose stage
    }
    for (int i = tid; i < 4096; i += 256)
        Wsh[(i >> 6) * LDA + (i & 63)] = (_Float16)w[i];
    __syncthreads();
    const int wave = tid >> 5, lane = tid & 31;
    const int rt = wave & 3, cg = wave >> 2;
    const v16h a0 = frag_a(Ash, rt * 16, 0, lane);
    const v16h a1 = frag_a(Ash, rt * 16, 32, lane);
    const int  nl = lane & 15;
    const int  mb = p0 + rt * 16 + ((lane >> 4) << 3);
#pragma unroll
    for (int q = 0; q < 2; ++q) {
        int ct = cg * 2 + q;
        v8f c = {};
        c = wmma32(a0, frag_b(Wsh, ct * 16, 0, lane),  c);
        c = wmma32(a1, frag_b(Wsh, ct * 16, 32, lane), c);
        int n = ct * 16 + nl;
        float bn = bias[n];
#pragma unroll
        for (int v = 0; v < 8; ++v)
            out[((long)b * 64 + n) * 20480 + mb + v] = c[v] + bn;
    }
}

// K7c: depthwise 5x5 (pad 2) + sigmoid, msf = ms*(1+attn)
__global__ void k7c_dw(const float* __restrict__ t, const float* __restrict__ ms,
                       const float* __restrict__ dww, const float* __restrict__ dwb,
                       float* __restrict__ msf) {
    long idx = (long)blockIdx.x * 256 + threadIdx.x;   // 2,621,440
    int j = (int)(idx % 160);
    long rest = idx / 160;
    int i = (int)(rest & 127); rest >>= 7;
    int c = (int)(rest & 63);
    int b = (int)(rest >> 6);
    const float* tp = t + ((long)(b * 64 + c)) * 20480;
    __builtin_prefetch(ms + idx, 0, 0);                // global_prefetch_b8
    float acc = dwb[c];
#pragma unroll
    for (int u = 0; u < 5; ++u) {
        int ii = i + u - 2;
        if (ii < 0 || ii > 127) continue;
#pragma unroll
        for (int v = 0; v < 5; ++v) {
            int jj = j + v - 2;
            if (jj < 0 || jj > 159) continue;
            acc += dww[c * 25 + u * 5 + v] * tp[ii * 160 + jj];
        }
    }
    msf[idx] = ms[idx] * (1.f + sigmf(acc));
}

// K8: out = (x_mamba + x_mask * shifted(msf)) @ W_toOut^T  (WMMA)
__global__ void k8_final(const float* __restrict__ xmamba, const float* __restrict__ xmask,
                         const float* __restrict__ msf, const float* __restrict__ w_out2,
                         float* __restrict__ out) {
    __shared__ __align__(16) _Float16 Ash[64 * LDA];
    __shared__ __align__(16) _Float16 Wsh[64 * LDA];
    const int tid = threadIdx.x;
    const int row0 = blockIdx.x * 64;
    const int b = row0 >> 14;
    for (int i = tid; i < 4096; i += 256) {
        int r = i >> 6, d = i & 63;
        int l = (row0 & 16383) + r;
        int ii = l >> 7, jj = l & 127;
        long g = (long)(row0 + r) * 64 + d;
        float ma = msf[(((long)(b * 64 + d)) * 128 + ii) * 160 + (d >> 1) + jj];
        Ash[r * LDA + d] = (_Float16)(xmamba[g] + xmask[g] * ma);
    }
    for (int i = tid; i < 4096; i += 256)
        Wsh[(i >> 6) * LDA + (i & 63)] = (_Float16)w_out2[i];
    __syncthreads();
    const int wave = tid >> 5, lane = tid & 31;
    const int rt = wave & 3, cg = wave >> 2;
    const v16h a0 = frag_a(Ash, rt * 16, 0, lane);
    const v16h a1 = frag_a(Ash, rt * 16, 32, lane);
    const int  nl = lane & 15;
    const long mbase = (long)row0 + rt * 16 + ((lane >> 4) << 3);
#pragma unroll
    for (int q = 0; q < 2; ++q) {
        int ct = cg * 2 + q;
        v8f c = {};
        c = wmma32(a0, frag_b(Wsh, ct * 16, 0, lane),  c);
        c = wmma32(a1, frag_b(Wsh, ct * 16, 32, lane), c);
        int n = ct * 16 + nl;
#pragma unroll
        for (int v = 0; v < 8; ++v) out[(mbase + v) * 64 + n] = c[v];
    }
}

// ===========================================================================
extern "C" void kernel_launch(void* const* d_in, const int* in_sizes, int n_in,
                              void* d_out, int out_size, void* d_ws, size_t ws_size,
                              hipStream_t stream) {
    (void)in_sizes; (void)n_in; (void)out_size; (void)ws_size;
    const float* x        = (const float*)d_in[0];
    const float* mask     = (const float*)d_in[1];
    const float* W_toMask = (const float*)d_in[2];
    const float* W_toOut  = (const float*)d_in[3];
    const float* mm_c1_w  = (const float*)d_in[4];
    const float* mm_c1_b  = (const float*)d_in[5];
    const float* mm_c2_w  = (const float*)d_in[6];
    const float* mm_c2_b  = (const float*)d_in[7];
    const float* mm_dw_w  = (const float*)d_in[8];
    const float* mm_dw_b  = (const float*)d_in[9];
    const float* m_in_w   = (const float*)d_in[10];
    const float* m_out_w  = (const float*)d_in[11];
    const float* m_norm_w = (const float*)d_in[12];
    const float* cw_f  = (const float*)d_in[13];
    const float* cb_f  = (const float*)d_in[14];
    const float* xp_f  = (const float*)d_in[15];
    const float* dtw_f = (const float*)d_in[16];
    const float* dtb_f = (const float*)d_in[17];
    const float* Al_f  = (const float*)d_in[18];
    const float* D_f   = (const float*)d_in[19];
    const float* cw_b  = (const float*)d_in[20];
    const float* cb_b  = (const float*)d_in[21];
    const float* xp_b  = (const float*)d_in[22];
    const float* dtw_b = (const float*)d_in[23];
    const float* dtb_b = (const float*)d_in[24];
    const float* Al_b  = (const float*)d_in[25];
    const float* D_b   = (const float*)d_in[26];

    float* ws = (float*)d_ws;
    float* xz     = ws;                       // 4,194,304
    float* xmaskb = ws + 4194304;             // 2,097,152
    float* xc     = ws + 6291456;             // 4,194,304
    float* dtbuf  = ws + 10485760;            // 4,194,304
    float* BCb    = ws + 14680064;            // 1,048,576  (B states 0..7, C states 8..15)
    float* wdt    = ws + 15728640;            // 8,192
    float* aggA   = ws + 15736832;            // 1,048,576
    float* aggB   = ws + 16785408;            // 1,048,576
    float* carry  = ws + 17833984;            // 1,048,576
    float* ybuf   = ws + 18882560;            // 4,194,304
    float* xmamba = ws + 23076864;            // 2,097,152
    float* msb    = ws + 25174016;            // 2,621,440
    float* tb     = ws + 27795456;            // 2,621,440
    float* msfb   = ws + 30416896;            // 2,621,440

    k1_inproj<<<512, 256, 0, stream>>>(x, m_in_w, W_toMask, xz, xmaskb);
    k0_wdt<<<32, 256, 0, stream>>>(dtw_f, xp_f, dtw_b, xp_b, wdt);
    k2a_conv<<<16384, 256, 0, stream>>>(xz, cw_f, cb_f, cw_b, cb_b, xc);
    k2b_proj<<<dim3(512, 2), 256, 0, stream>>>(xc, wdt, xp_f, xp_b, dtb_f, dtb_b,
                                               dtbuf, BCb);
    k3_scan1<<<dim3(NCHUNK, 2, 2), 64, 0, stream>>>(dtbuf, xc, BCb, Al_f, Al_b, aggA, aggB);
    k4_carry<<<8, 256, 0, stream>>>(aggA, aggB, carry);
    k5_scan2<<<dim3(NCHUNK, 2, 2), 64, 0, stream>>>(dtbuf, xc, BCb, xz, carry,
                                                    Al_f, Al_b, D_f, D_b, ybuf);
    k6_outproj<<<512, 256, 0, stream>>>(ybuf, m_norm_w, m_out_w, xmamba);
    k7_conv1x1<<<dim3(320, 2), 256, 0, stream>>>(mask, mm_c1_w, mm_c1_b, msb);
    k7_conv1x1<<<dim3(320, 2), 256, 0, stream>>>(msb, mm_c2_w, mm_c2_b, tb);
    k7c_dw<<<10240, 256, 0, stream>>>(tb, msb, mm_dw_w, mm_dw_b, msfb);
    k8_final<<<512, 256, 0, stream>>>(xmamba, xmaskb, msfb, W_toOut, (float*)d_out);
}
